// VectorQuantizer_9844065042629
// MI455X (gfx1250) — matrix operand order repaired and checked by reading, not compiled, not run on hardware
//
#include <hip/hip_runtime.h>
#include <hip/hip_bf16.h>
#include <stddef.h>

// Problem constants from the reference
#define N_ROWS   131072         // 8192*4*4
#define DIM      256            // feature dim
#define KCODES   1024           // codebook size
#define COL_TILES (KCODES/16)   // 64
#define D_CHUNKS  (DIM/32)      // 8 WMMA K-chunks of 32
#define TILE_BYTES (16*DIM*2)   // one B tile: 16 codes x 256 dims x bf16 = 8KB

typedef __attribute__((ext_vector_type(16))) __bf16 v16bf;
typedef __attribute__((ext_vector_type(8)))  float  v8f;
typedef __bf16 __attribute__((address_space(3))) lds_bf16;

// ---------------- workspace layout (bytes) ----------------
#define WS_EMBT_F32   0          // K x D fp32 transposed codebook (exact gather)
#define WS_EMBT_BF16  1048576    // K x D bf16 (WMMA B operand)
#define WS_ENORM      1572864    // K fp32 code norms
#define WS_IDX        1576960    // N int32 winning indices
#define WS_LOSS       2101248    // fp32 loss accumulator

// ---- CDNA5 async copy global->LDS (ASYNCcnt-tracked), 16 bytes per lane ----
__device__ __forceinline__ void async_copy_b128(unsigned lds_byte_off,
                                                const void* gptr) {
    asm volatile("global_load_async_to_lds_b128 %0, %1, off"
                 :: "v"(lds_byte_off), "v"(gptr) : "memory");
}

#if __has_builtin(__builtin_amdgcn_s_wait_asynccnt)
#define WAIT_ASYNC(n) __builtin_amdgcn_s_wait_asynccnt(n)
#else
#define WAIT_ASYNC(n) asm volatile("s_wait_asynccnt %0" :: "i"(n) : "memory")
#endif

// -------- kernel 1: transpose codebook, code norms, zero loss acc --------
__global__ void vq_prep_kernel(const float* __restrict__ emb,
                               float* __restrict__ embT_f32,
                               __bf16* __restrict__ embT_bf16,
                               float* __restrict__ e_norms,
                               float* __restrict__ loss_acc) {
    const int k = blockIdx.x;       // 0..1023
    const int d = threadIdx.x;      // 0..255
    float v = emb[(size_t)d * KCODES + k];
    embT_f32 [(size_t)k * DIM + d] = v;
    embT_bf16[(size_t)k * DIM + d] = (__bf16)v;

    __shared__ float s[256];
    s[d] = v * v;
    __syncthreads();
    #pragma unroll
    for (int off = 128; off > 0; off >>= 1) {
        if (d < off) s[d] += s[d + off];
        __syncthreads();
    }
    if (d == 0) e_norms[k] = s[0];
    if (k == 0 && d == 0) *loss_acc = 0.0f;
}

// -------- kernel 2: WMMA distance sweep + per-row argmin --------
// One wave32 owns a 16-row tile. All 8 waves in the block share B tiles
// staged in LDS by a double-buffered async-to-LDS pipeline; within a wave
// the B operand is additionally register-pipelined one chunk ahead so each
// WMMA never waits on the ds_load issued directly before it.
// dist(n,k) = ||e_k||^2 - 2*sim(n,k)   (||f_n||^2 dropped: row-constant)
__global__ void __launch_bounds__(256)
vq_argmin_kernel(const float* __restrict__ x,
                 const __bf16* __restrict__ embT,      // K x D bf16
                 const float* __restrict__ e_norms,    // K
                 int*   __restrict__ idx_i32,
                 float* __restrict__ idx_out_f) {
    __shared__ __align__(64) __bf16 Bs[2][16 * DIM];   // 2 x 8KB double buffer

    const int tid    = threadIdx.x;
    const int lane   = tid & 31;
    const int wave   = tid >> 5;                       // 0..7
    const int half   = lane >> 4;                      // 0 | 1
    const int mn     = lane & 15;                      // A: row M ; B/C: col N
    const int rowBase = (blockIdx.x * 8 + wave) * 16;

    // LDS byte offset of Bs (AS3 pointer -> 32-bit LDS address)
    const unsigned ldsBase = (unsigned)(size_t)(lds_bf16*)&Bs[0][0];

    // ---- load A tile (16 rows x 256 cols) as bf16 in contiguous k order (same order as B) ----
    // chunk dc covers K(d) in [dc*32, dc*32+32):
    //   elems 0..7  -> d = dc*32 + 16*half + 0..7
    //   elems 8..15 -> d = dc*32 + 16*half + 8 + 0..7
    v16bf A[D_CHUNKS];
    {
        const float* arow = x + (size_t)(rowBase + mn) * DIM;
        #pragma unroll
        for (int dc = 0; dc < D_CHUNKS; ++dc) {
            const float* p0 = arow + dc * 32 + 16 * half;
            const float* p1 = p0 + 8;
            #pragma unroll
            for (int j = 0; j < 8; ++j) {
                A[dc][j]     = (__bf16)p0[j];
                A[dc][8 + j] = (__bf16)p1[j];
            }
        }
    }

    float best[8];
    int   bestk[8];
    #pragma unroll
    for (int r = 0; r < 8; ++r) { best[r] = 3.4e38f; bestk[r] = 0; }

    // Stage issue: 256 threads x 32B = 8KB tile, 2 async b128 per thread.
    const char* embB = (const char*)embT;
    auto issue_stage = [&](int ct, int buf) {
        const char* g  = embB + (size_t)ct * TILE_BYTES + tid * 32;
        unsigned    ld = ldsBase + (unsigned)buf * TILE_BYTES + tid * 32;
        async_copy_b128(ld,      g);
        async_copy_b128(ld + 16, g + 16);
    };

    issue_stage(0, 0);

    for (int ct = 0; ct < COL_TILES; ++ct) {
        const int buf = ct & 1;
        const bool more = (ct + 1) < COL_TILES;
        if (more) issue_stage(ct + 1, buf ^ 1);
        // 2 async issues/stage/wave; in-order completion: <=2 outstanding
        // guarantees the *current* tile's copies have landed in LDS.
        if (more) { WAIT_ASYNC(2); } else { WAIT_ASYNC(0); }
        __syncthreads();   // all waves' copies visible before consuming

        const int kcol = ct * 16 + mn;                 // this lane's code id
        const float en = e_norms[kcol];

        v8f c = {};
        const __bf16* brow = &Bs[buf][mn * DIM + 16 * half];
        // B layout: lane col = mn; elems j -> K(d) = dc*32 + 16*half + j
        // Register-pipeline: fetch chunk dc+1 while WMMA consumes chunk dc.
        v16bf bcur = *(const v16bf*)(brow);
        #pragma unroll
        for (int dc = 0; dc < D_CHUNKS; ++dc) {
            v16bf bnext;
            if (dc + 1 < D_CHUNKS)
                bnext = *(const v16bf*)(brow + (dc + 1) * 32);
            c = __builtin_amdgcn_wmma_f32_16x16x32_bf16(
                    false, A[dc], false, bcur, (short)0, c, false, false);
            bcur = bnext;
        }
        // C layout: VGPR r holds M = r + 8*half, N = mn  (k fixed per lane)
        #pragma unroll
        for (int r = 0; r < 8; ++r) {
            float d = en - 2.0f * c[r];
            if (d < best[r]) { best[r] = d; bestk[r] = kcol; }
        }
        __syncthreads();   // done reading buf before it is overwritten
    }

    // ---- argmin across the 16 lanes sharing each row M ----
    #pragma unroll
    for (int r = 0; r < 8; ++r) {
        float b = best[r];
        int   k = bestk[r];
        #pragma unroll
        for (int mask = 1; mask < 16; mask <<= 1) {
            float ob = __shfl_xor(b, mask, 32);
            int   ok = __shfl_xor(k, mask, 32);
            if (ob < b || (ob == b && ok < k)) { b = ob; k = ok; }
        }
        if (mn == 0) {
            int row = rowBase + r + 8 * half;          // lanes 0 and 16 write
            idx_i32[row]   = k;
            idx_out_f[row] = (float)k;
        }
    }
}

// -------- kernel 3: fp32 gather of winning codes + loss partial --------
__global__ void vq_quant_kernel(const float* __restrict__ x,
                                const float* __restrict__ embT_f32,
                                const int* __restrict__ idx_i32,
                                float* __restrict__ qout,
                                float* __restrict__ loss_acc) {
    const size_t i4  = (size_t)blockIdx.x * blockDim.x + threadIdx.x; // float4 id
    const int    row = (int)(i4 >> 6);                 // 64 float4 per row
    const int    c4  = (int)(i4 & 63);
    const int    k   = idx_i32[row];

    float4 q  = ((const float4*)(embT_f32 + (size_t)k * DIM))[c4];
    float4 xv = ((const float4*)x)[i4];
    ((float4*)qout)[i4] = q;

    float dx = q.x - xv.x, dy = q.y - xv.y, dz = q.z - xv.z, dw = q.w - xv.w;
    float p = dx * dx + dy * dy + dz * dz + dw * dw;

    __shared__ float s[256];
    const int tid = threadIdx.x;
    s[tid] = p;
    __syncthreads();
    #pragma unroll
    for (int off = 128; off > 0; off >>= 1) {
        if (tid < off) s[tid] += s[tid + off];
        __syncthreads();
    }
    if (tid == 0) atomicAdd(loss_acc, s[0]);
}

// -------- kernel 4: finalize loss --------
__global__ void vq_fin_kernel(const float* __restrict__ loss_acc,
                              float* __restrict__ out_loss) {
    // vq_loss = (1 + BETA) * mean((q-x)^2), BETA = 0.25
    out_loss[0] = 1.25f * loss_acc[0] * (1.0f / (float)((size_t)N_ROWS * DIM));
}

extern "C" void kernel_launch(void* const* d_in, const int* in_sizes, int n_in,
                              void* d_out, int out_size, void* d_ws, size_t ws_size,
                              hipStream_t stream) {
    (void)in_sizes; (void)n_in; (void)out_size; (void)ws_size;

    const float* x   = (const float*)d_in[0];   // [8192,4,4,256] fp32
    const float* emb = (const float*)d_in[1];   // [256,1024] fp32

    char* ws = (char*)d_ws;
    float*  embT_f32  = (float*)(ws + WS_EMBT_F32);
    __bf16* embT_bf16 = (__bf16*)(ws + WS_EMBT_BF16);
    float*  e_norms   = (float*)(ws + WS_ENORM);
    int*    idx_i32   = (int*)(ws + WS_IDX);
    float*  loss_acc  = (float*)(ws + WS_LOSS);

    float* out_q    = (float*)d_out;                          // N*D
    float* out_idx  = out_q + (size_t)N_ROWS * DIM;           // N
    float* out_loss = out_idx + N_ROWS;                       // 1

    vq_prep_kernel<<<KCODES, DIM, 0, stream>>>(emb, embT_f32, embT_bf16,
                                               e_norms, loss_acc);

    // 8192 row tiles, 8 waves (tiles) per 256-thread block -> 1024 blocks
    vq_argmin_kernel<<<N_ROWS / 16 / 8, 256, 0, stream>>>(x, embT_bf16, e_norms,
                                                          idx_i32, out_idx);

    // 33554432 floats / 4 per thread / 256 threads = 32768 blocks
    vq_quant_kernel<<<(size_t)N_ROWS * DIM / 4 / 256, 256, 0, stream>>>(
        x, embT_f32, idx_i32, out_q, loss_acc);

    vq_fin_kernel<<<1, 1, 0, stream>>>(loss_acc, out_loss);
}
